// EDM_Block_75024488726872
// MI455X (gfx1250) — matrix operand order repaired and checked by reading, not compiled, not run on hardware
//
#include <hip/hip_runtime.h>
#include <hip/hip_bf16.h>

// ---------------------------------------------------------------------------
// EDM block for MI455X (gfx1250): all dense layers via v_wmma_f32_16x16x32_f16
// ---------------------------------------------------------------------------

#define E_TOT   400000
#define NN_TOT  50000
#define HDIM    128

typedef __attribute__((ext_vector_type(16))) _Float16 v16h;
typedef __attribute__((ext_vector_type(8)))  float    v8f;

struct Frag32 { uint4 lo; uint4 hi; };

static __device__ __forceinline__ v16h ldfrag(const _Float16* p0, const _Float16* p1) {
    Frag32 t;
    t.lo = *(const uint4*)p0;
    t.hi = *(const uint4*)p1;
    return __builtin_bit_cast(v16h, t);
}

static __device__ __forceinline__ void wbar() {
    // Same-wave LDS ops are processed in order by HW; this stops the compiler
    // from hoisting cross-lane LDS reads above the producing stores.
    __builtin_amdgcn_wave_barrier();
    asm volatile("" ::: "memory");
}

static __device__ __forceinline__ float silu(float x) {
    return x / (1.0f + __expf(-x));
}

static __device__ __forceinline__ v16h bfrag(const _Float16* wT, int kin, int n, int s, int khalf) {
    // B layout (KxN, 16-bit): lane n = l&15, Khalf = l>>4, K in {8*Khalf..+7} u {16+8*Khalf..+7}
    // wT is [Nout][Kin] row-major f16  ->  two 16B loads per lane.
    const _Float16* bp = wT + (size_t)n * kin + 32 * s + 8 * khalf;
    return ldfrag(bp, bp + 16);
}

static __device__ __forceinline__ v16h afrag(const _Float16* row, int s, int khalf) {
    // A layout (16x32, 16-bit): lane M = l&15, Khalf = l>>4,
    // K in {8*Khalf..+7} u {16+8*Khalf..+7}  -> two 16B LDS loads.
    const _Float16* ap = row + 32 * s + 8 * khalf;
    return ldfrag(ap, ap + 16);
}

#define WMMA(A, B, C) \
    __builtin_amdgcn_wmma_f32_16x16x32_f16(false, (A), false, (B), (short)0, (C), false, false)

// ---------------------------------------------------------------------------
// Prep: transpose fp32 weight [K][N] -> f16 [N][K]
// ---------------------------------------------------------------------------
__global__ void tr_f16_kernel(const float* __restrict__ src, _Float16* __restrict__ dst,
                              int K, int N) {
    int idx = blockIdx.x * blockDim.x + threadIdx.x;
    if (idx >= K * N) return;
    int k = idx / N;
    int n = idx - k * N;
    dst[(size_t)n * K + k] = (_Float16)src[(size_t)k * N + n];
}

// h0 = node_emb @ ne_w + ne_b   (N x 8) @ (8 x 128)
__global__ void node_embed_kernel(const float* __restrict__ ne, const float* __restrict__ w,
                                  const float* __restrict__ b, float* __restrict__ h) {
    int idx = blockIdx.x * blockDim.x + threadIdx.x;
    if (idx >= NN_TOT * HDIM) return;
    int n = idx >> 7;
    int j = idx & 127;
    float acc = b[j];
    const float* row = ne + (size_t)n * 8;
#pragma unroll
    for (int c = 0; c < 8; ++c) acc += row[c] * w[c * HDIM + j];
    h[idx] = acc;
}

__global__ void zero_kernel(float* __restrict__ p, int n) {
    int idx = blockIdx.x * blockDim.x + threadIdx.x;
    if (idx < n) p[idx] = 0.0f;
}

// ---------------------------------------------------------------------------
// Edge encoder: edge_attr = (relu(len*w1+b1) @ w2 + b2) * bond_emb[type]
// ---------------------------------------------------------------------------
__global__ __launch_bounds__(128) void edge_enc_kernel(
    const float* __restrict__ elen, const int* __restrict__ etype,
    const float* __restrict__ w1, const float* __restrict__ b1,
    const _Float16* __restrict__ w2T, const float* __restrict__ b2,
    const float* __restrict__ bond, _Float16* __restrict__ eattr) {
    __shared__ __align__(16) _Float16 smem[4 * 16 * 128];
    const int wid = threadIdx.x >> 5, lane = threadIdx.x & 31;
    const int tile = blockIdx.x * 4 + wid;
    const int tbase = tile * 16;
    const int mrow = lane & 15, khalf = lane >> 4;
    _Float16* aT = smem + wid * 16 * 128;

    for (int e = 0; e < 16; ++e) {
        float len = elen[tbase + e];
        for (int f = lane; f < HDIM; f += 32) {
            float x = fmaf(len, w1[f], b1[f]);
            aT[e * 128 + f] = (_Float16)fmaxf(x, 0.0f);
        }
    }
    wbar();

    v8f acc[8] = {};
#pragma unroll
    for (int s = 0; s < 4; ++s) {
        v16h a = afrag(aT + mrow * 128, s, khalf);
#pragma unroll
        for (int nt = 0; nt < 8; ++nt)
            acc[nt] = WMMA(a, bfrag(w2T, 128, nt * 16 + mrow, s, khalf), acc[nt]);
    }

    int et[8];
#pragma unroll
    for (int r = 0; r < 8; ++r) et[r] = etype[tbase + r + 8 * khalf];
#pragma unroll
    for (int nt = 0; nt < 8; ++nt) {
        const int n = nt * 16 + mrow;
        const float bias = b2[n];
#pragma unroll
        for (int r = 0; r < 8; ++r) {
            const int e = tbase + r + 8 * khalf;
            float v = (acc[nt][r] + bias) * bond[(size_t)et[r] * HDIM + n];
            eattr[(size_t)e * HDIM + n] = (_Float16)v;
        }
    }
}

// ---------------------------------------------------------------------------
// Conv edge: m = silu(silu([h_r|h_c|ea] @ w1 + b1) @ w2 + b2); agg[row] += m
// ---------------------------------------------------------------------------
__global__ __launch_bounds__(128) void conv_edge_kernel(
    const float* __restrict__ h, const _Float16* __restrict__ eattr,
    const int* __restrict__ eidx,
    const _Float16* __restrict__ w1T, const float* __restrict__ b1,
    const _Float16* __restrict__ w2T, const float* __restrict__ b2,
    float* __restrict__ agg) {
    __shared__ __align__(16) _Float16 smem[4 * 16 * 384];
    const int wid = threadIdx.x >> 5, lane = threadIdx.x & 31;
    const int tile = blockIdx.x * 4 + wid;
    const int tbase = tile * 16;
    const int mrow = lane & 15, khalf = lane >> 4;
    _Float16* aT = smem + wid * 16 * 384;

    for (int e = 0; e < 16; ++e) {
        const int ge = tbase + e;
        const float* hr = h + (size_t)eidx[ge] * HDIM;
        const float* hc = h + (size_t)eidx[E_TOT + ge] * HDIM;
        const _Float16* ea = eattr + (size_t)ge * HDIM;
        for (int f = lane; f < HDIM; f += 32) {
            aT[e * 384 + f]       = (_Float16)hr[f];
            aT[e * 384 + 128 + f] = (_Float16)hc[f];
            aT[e * 384 + 256 + f] = ea[f];
        }
    }
    wbar();

    v8f acc[8] = {};
#pragma unroll
    for (int s = 0; s < 12; ++s) {
        v16h a = afrag(aT + mrow * 384, s, khalf);
#pragma unroll
        for (int nt = 0; nt < 8; ++nt)
            acc[nt] = WMMA(a, bfrag(w1T, 384, nt * 16 + mrow, s, khalf), acc[nt]);
    }
#pragma unroll
    for (int nt = 0; nt < 8; ++nt) {
        const int n = nt * 16 + mrow;
        const float bias = b1[n];
#pragma unroll
        for (int r = 0; r < 8; ++r)
            aT[(r + 8 * khalf) * 384 + n] = (_Float16)silu(acc[nt][r] + bias);
    }
    wbar();

    v8f acc2[8] = {};
#pragma unroll
    for (int s = 0; s < 4; ++s) {
        v16h a = afrag(aT + mrow * 384, s, khalf);
#pragma unroll
        for (int nt = 0; nt < 8; ++nt)
            acc2[nt] = WMMA(a, bfrag(w2T, 128, nt * 16 + mrow, s, khalf), acc2[nt]);
    }
    int rows[8];
#pragma unroll
    for (int r = 0; r < 8; ++r) rows[r] = eidx[tbase + r + 8 * khalf];
#pragma unroll
    for (int nt = 0; nt < 8; ++nt) {
        const int n = nt * 16 + mrow;
        const float bias = b2[n];
#pragma unroll
        for (int r = 0; r < 8; ++r)
            atomicAdd(&agg[(size_t)rows[r] * HDIM + n], silu(acc2[nt][r] + bias));
    }
}

// ---------------------------------------------------------------------------
// Conv node: h += silu([h|agg/100] @ w1 + b1) @ w2 + b2
// ---------------------------------------------------------------------------
__global__ __launch_bounds__(128) void conv_node_kernel(
    float* __restrict__ h, const float* __restrict__ agg,
    const _Float16* __restrict__ w1T, const float* __restrict__ b1,
    const _Float16* __restrict__ w2T, const float* __restrict__ b2) {
    __shared__ __align__(16) _Float16 smem[4 * 16 * 256];
    const int wid = threadIdx.x >> 5, lane = threadIdx.x & 31;
    const int tile = blockIdx.x * 4 + wid;
    if (tile >= NN_TOT / 16) return;  // uniform per wave, no block barriers used
    const int tbase = tile * 16;
    const int mrow = lane & 15, khalf = lane >> 4;
    _Float16* aT = smem + wid * 16 * 256;

    for (int e = 0; e < 16; ++e) {
        const int n = tbase + e;
        for (int f = lane; f < HDIM; f += 32) {
            aT[e * 256 + f]       = (_Float16)h[(size_t)n * HDIM + f];
            aT[e * 256 + 128 + f] = (_Float16)(agg[(size_t)n * HDIM + f] * 0.01f);
        }
    }
    wbar();

    v8f acc[8] = {};
#pragma unroll
    for (int s = 0; s < 8; ++s) {
        v16h a = afrag(aT + mrow * 256, s, khalf);
#pragma unroll
        for (int nt = 0; nt < 8; ++nt)
            acc[nt] = WMMA(a, bfrag(w1T, 256, nt * 16 + mrow, s, khalf), acc[nt]);
    }
#pragma unroll
    for (int nt = 0; nt < 8; ++nt) {
        const int n = nt * 16 + mrow;
        const float bias = b1[n];
#pragma unroll
        for (int r = 0; r < 8; ++r)
            aT[(r + 8 * khalf) * 256 + n] = (_Float16)silu(acc[nt][r] + bias);
    }
    wbar();

    v8f acc2[8] = {};
#pragma unroll
    for (int s = 0; s < 4; ++s) {
        v16h a = afrag(aT + mrow * 256, s, khalf);
#pragma unroll
        for (int nt = 0; nt < 8; ++nt)
            acc2[nt] = WMMA(a, bfrag(w2T, 128, nt * 16 + mrow, s, khalf), acc2[nt]);
    }
#pragma unroll
    for (int nt = 0; nt < 8; ++nt) {
        const int n = nt * 16 + mrow;
        const float bias = b2[n];
#pragma unroll
        for (int r = 0; r < 8; ++r) {
            const size_t o = (size_t)(tbase + r + 8 * khalf) * HDIM + n;
            h[o] = h[o] + acc2[nt][r] + bias;  // residual
        }
    }
}

// ---------------------------------------------------------------------------
// Grad MLP + position scatter
// ---------------------------------------------------------------------------
__global__ __launch_bounds__(128) void grad_kernel(
    const float* __restrict__ h, const _Float16* __restrict__ eattr,
    const int* __restrict__ eidx, const float* __restrict__ elen,
    const float* __restrict__ pos,
    const _Float16* __restrict__ w1T, const float* __restrict__ b1,
    const _Float16* __restrict__ w2T, const float* __restrict__ b2,
    const float* __restrict__ w3, const float* __restrict__ b3,
    float* __restrict__ posagg) {
    __shared__ __align__(16) _Float16 smem[4 * 16 * 384];
    const int wid = threadIdx.x >> 5, lane = threadIdx.x & 31;
    const int tile = blockIdx.x * 4 + wid;
    const int tbase = tile * 16;
    const int mrow = lane & 15, khalf = lane >> 4;
    _Float16* aT = smem + wid * 16 * 384;

    for (int e = 0; e < 16; ++e) {
        const int ge = tbase + e;
        const float* hr = h + (size_t)eidx[ge] * HDIM;
        const float* hc = h + (size_t)eidx[E_TOT + ge] * HDIM;
        const _Float16* ea = eattr + (size_t)ge * HDIM;
        for (int f = lane; f < HDIM; f += 32) {
            aT[e * 384 + f]       = (_Float16)hr[f];
            aT[e * 384 + 128 + f] = (_Float16)hc[f];
            aT[e * 384 + 256 + f] = ea[f];
        }
    }
    wbar();

    v8f acc[8] = {};
#pragma unroll
    for (int s = 0; s < 12; ++s) {
        v16h a = afrag(aT + mrow * 384, s, khalf);
#pragma unroll
        for (int nt = 0; nt < 8; ++nt)
            acc[nt] = WMMA(a, bfrag(w1T, 384, nt * 16 + mrow, s, khalf), acc[nt]);
    }
#pragma unroll
    for (int nt = 0; nt < 8; ++nt) {
        const int n = nt * 16 + mrow;
        const float bias = b1[n];
#pragma unroll
        for (int r = 0; r < 8; ++r)
            aT[(r + 8 * khalf) * 384 + n] = (_Float16)fmaxf(acc[nt][r] + bias, 0.0f);
    }
    wbar();

    v8f acc2[4] = {};  // 128 -> 64
#pragma unroll
    for (int s = 0; s < 4; ++s) {
        v16h a = afrag(aT + mrow * 384, s, khalf);
#pragma unroll
        for (int nt = 0; nt < 4; ++nt)
            acc2[nt] = WMMA(a, bfrag(w2T, 128, nt * 16 + mrow, s, khalf), acc2[nt]);
    }
#pragma unroll
    for (int nt = 0; nt < 4; ++nt) {
        const int n = nt * 16 + mrow;
        const float bias = b2[n];
#pragma unroll
        for (int r = 0; r < 8; ++r)
            aT[(r + 8 * khalf) * 384 + 256 + n] = (_Float16)fmaxf(acc2[nt][r] + bias, 0.0f);
    }
    wbar();

    if (lane < 16) {
        const int e = tbase + lane;
        float s = b3[0];
#pragma unroll 8
        for (int k = 0; k < 64; ++k)
            s += (float)aT[lane * 384 + 256 + k] * w3[k];
        const int r = eidx[e], c = eidx[E_TOT + e];
        const float il = 1.0f / elen[e];
        const float dx = (pos[2 * r] - pos[2 * c]) * il;
        const float dy = (pos[2 * r + 1] - pos[2 * c + 1]) * il;
        atomicAdd(&posagg[2 * r],     dx * s);
        atomicAdd(&posagg[2 * r + 1], dy * s);
    }
}

__global__ void finalize_kernel(const float* __restrict__ pos, const float* __restrict__ pagg,
                                float* __restrict__ out, int n) {
    int idx = blockIdx.x * blockDim.x + threadIdx.x;
    if (idx < n) out[idx] = pos[idx] + pagg[idx] * 0.01f;
}

// ---------------------------------------------------------------------------
// Workspace layout (bytes)
// ---------------------------------------------------------------------------
#define OFF_EE2T   0u                      // 128x128 f16
#define OFF_CONVW  32768u                  // 4 convs
#define CONV_STRIDE 229376u                //  e_w1T(98304) e_w2T(32768) n_w1T(65536) n_w2T(32768)
#define OFF_GW1T   950272u
#define OFF_GW2T   1048576u
#define OFF_EATTR  1064960u                // E*128 f16
#define OFF_H      103464960u              // N*128 f32
#define OFF_AGG    129064960u              // N*128 f32
#define OFF_PAGG   154664960u              // N*2 f32

extern "C" void kernel_launch(void* const* d_in, const int* in_sizes, int n_in,
                              void* d_out, int out_size, void* d_ws, size_t ws_size,
                              hipStream_t stream) {
    const float* node_emb = (const float*)d_in[0];
    const float* pos      = (const float*)d_in[3];
    const int*   eidx     = (const int*)d_in[4];
    const int*   etype    = (const int*)d_in[5];
    const float* elen     = (const float*)d_in[6];
    const float* bond     = (const float*)d_in[9];
    const float* ee_w1    = (const float*)d_in[10];
    const float* ee_b1    = (const float*)d_in[11];
    const float* ee_w2    = (const float*)d_in[12];
    const float* ee_b2    = (const float*)d_in[13];
    const float* ne_w     = (const float*)d_in[14];
    const float* ne_b     = (const float*)d_in[15];
    const float* g_w1     = (const float*)d_in[16];
    const float* g_b1     = (const float*)d_in[17];
    const float* g_w2     = (const float*)d_in[18];
    const float* g_b2     = (const float*)d_in[19];
    const float* g_w3     = (const float*)d_in[20];
    const float* g_b3     = (const float*)d_in[21];

    char* ws = (char*)d_ws;
    _Float16* wEE2T = (_Float16*)(ws + OFF_EE2T);
    _Float16* wG1T  = (_Float16*)(ws + OFF_GW1T);
    _Float16* wG2T  = (_Float16*)(ws + OFF_GW2T);
    _Float16* eattr = (_Float16*)(ws + OFF_EATTR);
    float*    hbuf  = (float*)(ws + OFF_H);
    float*    agg   = (float*)(ws + OFF_AGG);
    float*    pagg  = (float*)(ws + OFF_PAGG);

    auto tr = [&](const float* src, _Float16* dst, int K, int N) {
        tr_f16_kernel<<<(K * N + 255) / 256, 256, 0, stream>>>(src, dst, K, N);
    };

    // Pack weights as f16 [out][in]
    tr(ee_w2, wEE2T, 128, 128);
    tr(g_w1, wG1T, 384, 128);
    tr(g_w2, wG2T, 128, 64);
    _Float16* cw1T[4]; _Float16* cw2T[4]; _Float16* nw1T[4]; _Float16* nw2T[4];
    const float *ceb1[4], *ceb2[4], *cnb1[4], *cnb2[4];
    for (int i = 0; i < 4; ++i) {
        char* base = ws + OFF_CONVW + (unsigned)i * CONV_STRIDE;
        cw1T[i] = (_Float16*)(base);
        cw2T[i] = (_Float16*)(base + 98304);
        nw1T[i] = (_Float16*)(base + 131072);
        nw2T[i] = (_Float16*)(base + 196608);
        const int b = 22 + 8 * i;
        tr((const float*)d_in[b + 0], cw1T[i], 384, 128);
        tr((const float*)d_in[b + 2], cw2T[i], 128, 128);
        tr((const float*)d_in[b + 4], nw1T[i], 256, 128);
        tr((const float*)d_in[b + 6], nw2T[i], 128, 128);
        ceb1[i] = (const float*)d_in[b + 1];
        ceb2[i] = (const float*)d_in[b + 3];
        cnb1[i] = (const float*)d_in[b + 5];
        cnb2[i] = (const float*)d_in[b + 7];
    }

    // h0 = node_emb @ ne_w + ne_b
    node_embed_kernel<<<(NN_TOT * HDIM + 255) / 256, 256, 0, stream>>>(node_emb, ne_w, ne_b, hbuf);

    // edge_attr
    edge_enc_kernel<<<E_TOT / 16 / 4, 128, 0, stream>>>(elen, etype, ee_w1, ee_b1,
                                                        wEE2T, ee_b2, bond, eattr);

    // 4 GCL layers
    for (int i = 0; i < 4; ++i) {
        zero_kernel<<<(NN_TOT * HDIM + 255) / 256, 256, 0, stream>>>(agg, NN_TOT * HDIM);
        conv_edge_kernel<<<E_TOT / 16 / 4, 128, 0, stream>>>(hbuf, eattr, eidx,
                                                             cw1T[i], ceb1[i], cw2T[i], ceb2[i], agg);
        conv_node_kernel<<<(NN_TOT / 16 + 3) / 4, 128, 0, stream>>>(hbuf, agg,
                                                                    nw1T[i], cnb1[i], nw2T[i], cnb2[i]);
    }

    // grad MLP + position update
    zero_kernel<<<(NN_TOT * 2 + 255) / 256, 256, 0, stream>>>(pagg, NN_TOT * 2);
    grad_kernel<<<E_TOT / 16 / 4, 128, 0, stream>>>(hbuf, eattr, eidx, elen, pos,
                                                    wG1T, g_b1, wG2T, g_b2, g_w3, g_b3, pagg);
    finalize_kernel<<<(NN_TOT * 2 + 255) / 256, 256, 0, stream>>>(pos, pagg, (float*)d_out,
                                                                  NN_TOT * 2);
    (void)in_sizes; (void)n_in; (void)out_size; (void)ws_size;
}
